// ClusterAttn_15109694947342
// MI455X (gfx1250) — matrix operand order repaired and checked by reading, not compiled, not run on hardware
//
#include <hip/hip_runtime.h>
#include <hip/hip_bf16.h>

typedef __attribute__((ext_vector_type(16))) _Float16 v16h;
typedef __attribute__((ext_vector_type(8)))  float    v8f;

#define DIMC 96
#define NTOK 8192
#define FEAD 64
#define KCL  64

// ---------------- WMMA helpers (wave32, this file's contiguous-k layouts) ----------------
// A 16x32 f16: lane L -> M = L%16; VGPR pair p, elem o:
//   K = (L<16 ? 0 : 16) + 2*p + o
__device__ __forceinline__ v16h load_A_h(const _Float16* tile, int ld) {
  int lane = threadIdx.x & 31;
  const _Float16* row = tile + (lane & 15) * ld;
  int kh = (lane < 16) ? 0 : 16;
  v16h a;
#pragma unroll
  for (int p = 0; p < 8; ++p) {
    int k = kh + 2 * p;
    a[2 * p]     = row[k];
    a[2 * p + 1] = row[k + 1];
  }
  return a;
}

// B 32x16 f16 (row-major K x N tile): lane L -> N = L%16;
//   elem e -> K = 16*(L/16) + e
__device__ __forceinline__ v16h load_B_h(const _Float16* tile, int ld) {
  int lane = threadIdx.x & 31;
  int n = lane & 15;
  int k0 = (lane < 16) ? 0 : 16;
  v16h b;
#pragma unroll
  for (int e = 0; e < 16; ++e) b[e] = tile[(k0 + e) * ld + n];
  return b;
}

// B[k][n] = W[n][k] from fp32 weight matrix (converted on load).
// W pre-advanced to (ntile*16)*ldw + ktile.
__device__ __forceinline__ v16h load_BT_f32(const float* W, int ldw) {
  int lane = threadIdx.x & 31;
  int n = lane & 15;
  int k0 = (lane < 16) ? 0 : 16;
  v16h b;
#pragma unroll
  for (int e = 0; e < 16; ++e) b[e] = (_Float16)W[n * ldw + k0 + e];
  return b;
}

__device__ __forceinline__ v8f wmma_f16(v16h a, v16h b, v8f c) {
  return __builtin_amdgcn_wmma_f32_16x16x32_f16(false, a, false, b, (short)0, c,
                                                false, false);
}

// ---------------- Stage 1: dwc conv (96ch -> 1ch, 3x3x3) ----------------
__global__ void k_dwc(const float* __restrict__ x, const float* __restrict__ w,
                      const float* __restrict__ bias, float* __restrict__ fea) {
  __shared__ float wl[DIMC * 27];
  for (int i = threadIdx.x; i < DIMC * 27; i += blockDim.x) wl[i] = w[i];
  __syncthreads();
  int v = blockIdx.x * blockDim.x + threadIdx.x;  // 0..524287
  int z = v >> 14, y = (v >> 7) & 127, xx = v & 127;
  float a0 = bias[0], a1 = 0.f, a2 = 0.f, a3 = 0.f;
  for (int dz = -1; dz <= 1; ++dz)
    for (int dy = -1; dy <= 1; ++dy)
      for (int dx = -1; dx <= 1; ++dx) {
        int zz = z + dz, yy = y + dy, xc = xx + dx;
        if ((unsigned)zz >= 32u || (unsigned)yy >= 128u || (unsigned)xc >= 128u)
          continue;
        int t = (dz + 1) * 9 + (dy + 1) * 3 + (dx + 1);
        const float4* px = reinterpret_cast<const float4*>(
            x + (((size_t)zz * 128 + yy) * 128 + xc) * DIMC);
#pragma unroll
        for (int j = 0; j < DIMC / 4; ++j) {
          float4 p = px[j];
          int c = 4 * j;
          a0 += p.x * wl[c * 27 + t];
          a1 += p.y * wl[(c + 1) * 27 + t];
          a2 += p.z * wl[(c + 2) * 27 + t];
          a3 += p.w * wl[(c + 3) * 27 + t];
        }
      }
  float acc = (a0 + a1) + (a2 + a3);
  int zb = z >> 2, zp = z & 3, yb = y >> 2, yp = y & 3, xb = xx >> 2, xp = xx & 3;
  int n = (zb * 32 + yb) * 32 + xb;
  int d = (zp * 4 + yp) * 4 + xp;
  fea[n * FEAD + d] = acc;
}

// ---------------- Stage 2: column norms + normalize to f16 --------------
__global__ void k_colnorm(const float* __restrict__ fea, float* __restrict__ cn) {
  __shared__ float red[256];
  int d = blockIdx.x;
  float s = 0.f;
  for (int n = threadIdx.x; n < NTOK; n += 256) {
    float f = fea[n * FEAD + d];
    s += f * f;
  }
  red[threadIdx.x] = s;
  __syncthreads();
  for (int o = 128; o > 0; o >>= 1) {
    if (threadIdx.x < o) red[threadIdx.x] += red[threadIdx.x + o];
    __syncthreads();
  }
  if (threadIdx.x == 0) cn[d] = sqrtf(red[0]);
}

__global__ void k_feanorm(const float* __restrict__ fea,
                          const float* __restrict__ cn,
                          _Float16* __restrict__ fh) {
  int i = blockIdx.x * blockDim.x + threadIdx.x;
  int d = i & 63;
  float nv = fmaxf(cn[d], 1e-12f);
  fh[i] = (_Float16)(fea[i] / nv);
}

// ---------------- Stage 3: logits = fea @ cent^T (WMMA) -----------------
__global__ void k_logits(const _Float16* __restrict__ feah,
                         const float* __restrict__ cent,
                         float* __restrict__ logits) {
  int wid = threadIdx.x >> 5;
  int mtile = blockIdx.x * 8 + wid;
  int ntile = blockIdx.y;
  v8f acc = {};
  const _Float16* abase = feah + mtile * 16 * FEAD;
#pragma unroll
  for (int kt = 0; kt < FEAD; kt += 32) {
    v16h a = load_A_h(abase + kt, FEAD);
    v16h b = load_BT_f32(cent + ntile * 16 * FEAD + kt, FEAD);
    acc = wmma_f16(a, b, acc);
  }
  int lane = threadIdx.x & 31;
  int ncol = ntile * 16 + (lane & 15);
  int mrow0 = mtile * 16 + ((lane < 16) ? 0 : 8);
#pragma unroll
  for (int r = 0; r < 8; ++r) logits[(mrow0 + r) * KCL + ncol] = acc[r];
}

// ---------------- Stage 3b: column softmax stats over N ------------------
__global__ void k_colmaxsum(const float* __restrict__ logits,
                            float* __restrict__ cmax, float* __restrict__ csum) {
  __shared__ float red[256];
  int c = blockIdx.x;
  float m = -3.4e38f;
  for (int n = threadIdx.x; n < NTOK; n += 256)
    m = fmaxf(m, logits[n * KCL + c]);
  red[threadIdx.x] = m;
  __syncthreads();
  for (int o = 128; o > 0; o >>= 1) {
    if (threadIdx.x < o) red[threadIdx.x] = fmaxf(red[threadIdx.x], red[threadIdx.x + o]);
    __syncthreads();
  }
  float mx = red[0];
  __syncthreads();
  float s = 0.f;
  for (int n = threadIdx.x; n < NTOK; n += 256)
    s += __expf(logits[n * KCL + c] - mx);
  red[threadIdx.x] = s;
  __syncthreads();
  for (int o = 128; o > 0; o >>= 1) {
    if (threadIdx.x < o) red[threadIdx.x] += red[threadIdx.x + o];
    __syncthreads();
  }
  if (threadIdx.x == 0) { cmax[c] = mx; csum[c] = red[0]; }
}

// Et[c][n] = exp(logits[n][c] - cmax[c])  (WMMA-A-ready transposed layout)
__global__ void k_expT(const float* __restrict__ logits,
                       const float* __restrict__ cmax, _Float16* __restrict__ Et) {
  int i = blockIdx.x * blockDim.x + threadIdx.x;
  int c = i >> 13, n = i & (NTOK - 1);
  Et[i] = (_Float16)__expf(logits[n * KCL + c] - cmax[c]);
}

// ---------------- Stage 3c: new_centroids = softmax^T @ fea (WMMA) ------
__global__ void k_newcent(const _Float16* __restrict__ Et,
                          const _Float16* __restrict__ feah,
                          const float* __restrict__ csum, float* __restrict__ nc) {
  int wid = threadIdx.x >> 5;  // 0..15
  int mtile = wid >> 2, ntile = wid & 3;
  v8f acc = {};
  for (int kt = 0; kt < NTOK; kt += 32) {
    v16h a = load_A_h(Et + mtile * 16 * NTOK + kt, NTOK);
    v16h b = load_B_h(feah + kt * FEAD + ntile * 16, FEAD);
    acc = wmma_f16(a, b, acc);
  }
  int lane = threadIdx.x & 31;
  int ncol = ntile * 16 + (lane & 15);
  int mrow0 = mtile * 16 + ((lane < 16) ? 0 : 8);
#pragma unroll
  for (int r = 0; r < 8; ++r) {
    int c = mrow0 + r;
    nc[c * FEAD + ncol] = acc[r] / csum[c];
  }
}

// ---------------- Stage 4: kv projection (tiny, VALU) -------------------
__global__ void k_kv(const float* __restrict__ nc, const float* __restrict__ kvw,
                     const float* __restrict__ kvb, _Float16* __restrict__ kth,
                     _Float16* __restrict__ vh) {
  for (int i = threadIdx.x; i < KCL * 2 * FEAD; i += blockDim.x) {
    int c = i >> 7, j = i & 127;
    float s = kvb[j];
    const float* row = kvw + j * FEAD;
    const float* nr = nc + c * FEAD;
#pragma unroll 8
    for (int e = 0; e < FEAD; ++e) s += nr[e] * row[e];
    if (j < FEAD)
      kth[j * KCL + c] = (_Float16)s;       // k^T: [d][c], B-ready for q@k^T
    else
      vh[c * FEAD + (j - FEAD)] = (_Float16)s;  // v: [c][d], B-ready for attn@v
  }
}

// ---------------- Stage 4b: q = fea @ q_w^T + q_b (WMMA) ----------------
__global__ void k_q(const _Float16* __restrict__ feah, const float* __restrict__ qw,
                    const float* __restrict__ qb, _Float16* __restrict__ qh) {
  int wid = threadIdx.x >> 5;
  int mtile = blockIdx.x * 8 + wid;
  int ntile = blockIdx.y;
  v8f acc = {};
  const _Float16* abase = feah + mtile * 16 * FEAD;
#pragma unroll
  for (int kt = 0; kt < FEAD; kt += 32) {
    v16h a = load_A_h(abase + kt, FEAD);
    v16h b = load_BT_f32(qw + ntile * 16 * FEAD + kt, FEAD);
    acc = wmma_f16(a, b, acc);
  }
  int lane = threadIdx.x & 31;
  int ncol = ntile * 16 + (lane & 15);
  int mrow0 = mtile * 16 + ((lane < 16) ? 0 : 8);
#pragma unroll
  for (int r = 0; r < 8; ++r)
    qh[(mrow0 + r) * FEAD + ncol] = (_Float16)(acc[r] + qb[ncol]);
}

// ---------------- Stage 5: fused attention (WMMA + softmax + WMMA) ------
__global__ void k_attn(const _Float16* __restrict__ qh,
                       const _Float16* __restrict__ kth,
                       const _Float16* __restrict__ vh, float* __restrict__ o) {
  __shared__ _Float16 sm[8 * 16 * 64];  // per-wave 16x64 attn tile
  int wid = threadIdx.x >> 5, lane = threadIdx.x & 31;
  int mtile = blockIdx.x * 8 + wid;

  v16h a0 = load_A_h(qh + mtile * 16 * FEAD + 0, FEAD);
  v16h a1 = load_A_h(qh + mtile * 16 * FEAD + 32, FEAD);
  v8f S[4];
#pragma unroll
  for (int nt = 0; nt < 4; ++nt) {
    v8f acc = {};
    acc = wmma_f16(a0, load_B_h(kth + 0 * KCL + nt * 16, KCL), acc);
    acc = wmma_f16(a1, load_B_h(kth + 32 * KCL + nt * 16, KCL), acc);
    S[nt] = acc;
  }
  // row softmax over 64 clusters (scale 1/sqrt(64) = 0.125)
#pragma unroll
  for (int r = 0; r < 8; ++r) {
    float mx = -3.4e38f;
#pragma unroll
    for (int nt = 0; nt < 4; ++nt) {
      S[nt][r] *= 0.125f;
      mx = fmaxf(mx, S[nt][r]);
    }
    mx = fmaxf(mx, __shfl_xor(mx, 1));
    mx = fmaxf(mx, __shfl_xor(mx, 2));
    mx = fmaxf(mx, __shfl_xor(mx, 4));
    mx = fmaxf(mx, __shfl_xor(mx, 8));
    float sum = 0.f;
#pragma unroll
    for (int nt = 0; nt < 4; ++nt) {
      S[nt][r] = __expf(S[nt][r] - mx);
      sum += S[nt][r];
    }
    sum += __shfl_xor(sum, 1);
    sum += __shfl_xor(sum, 2);
    sum += __shfl_xor(sum, 4);
    sum += __shfl_xor(sum, 8);
    float inv = 1.f / sum;
#pragma unroll
    for (int nt = 0; nt < 4; ++nt) S[nt][r] *= inv;
  }
  // C-layout -> A-layout via LDS round trip
  _Float16* my = sm + wid * 16 * 64;
  int m0 = (lane < 16) ? 0 : 8;
  int ncol = lane & 15;
#pragma unroll
  for (int nt = 0; nt < 4; ++nt)
#pragma unroll
    for (int r = 0; r < 8; ++r)
      my[(m0 + r) * 64 + nt * 16 + ncol] = (_Float16)S[nt][r];
  __syncthreads();
  v16h p0 = load_A_h(my + 0, 64);
  v16h p1 = load_A_h(my + 32, 64);
#pragma unroll
  for (int nt = 0; nt < 4; ++nt) {
    v8f acc = {};
    acc = wmma_f16(p0, load_B_h(vh + 0 * FEAD + nt * 16, FEAD), acc);
    acc = wmma_f16(p1, load_B_h(vh + 32 * FEAD + nt * 16, FEAD), acc);
    int tok0 = mtile * 16 + m0;
    int d = nt * 16 + ncol;
    int zp = d >> 4, yp = (d >> 2) & 3, xp = d & 3;
#pragma unroll
    for (int r = 0; r < 8; ++r) {
      int n = tok0 + r;
      int zb = n >> 10, yb = (n >> 5) & 31, xb = n & 31;
      int z = zb * 4 + zp, y = yb * 4 + yp, xx = xb * 4 + xp;
      o[((z * 128) + y) * 128 + xx] = acc[r];
    }
  }
}

// ---------------- Stage 6: upc conv (1ch -> 96ch) + residual ------------
__global__ void k_upc(const float* __restrict__ o, const float* __restrict__ w,
                      const float* __restrict__ bias, const float* __restrict__ x,
                      float* __restrict__ out) {
  __shared__ float wl[DIMC * 27];
  __shared__ float bl[DIMC];
  for (int i = threadIdx.x; i < DIMC * 27; i += blockDim.x) wl[i] = w[i];
  for (int i = threadIdx.x; i < DIMC; i += blockDim.x) bl[i] = bias[i];
  __syncthreads();
  int v = blockIdx.x * blockDim.x + threadIdx.x;
  int z = v >> 14, y = (v >> 7) & 127, xx = v & 127;
  float ov[27];
#pragma unroll
  for (int dz = -1; dz <= 1; ++dz)
#pragma unroll
    for (int dy = -1; dy <= 1; ++dy)
#pragma unroll
      for (int dx = -1; dx <= 1; ++dx) {
        int zz = z + dz, yy = y + dy, xc = xx + dx;
        int t = (dz + 1) * 9 + (dy + 1) * 3 + (dx + 1);
        ov[t] = ((unsigned)zz < 32u && (unsigned)yy < 128u && (unsigned)xc < 128u)
                    ? o[((zz * 128) + yy) * 128 + xc]
                    : 0.f;
      }
  size_t base = (size_t)v * DIMC;
  for (int c = 0; c < DIMC; ++c) {
    float s = bl[c];
    const float* wc = wl + c * 27;
#pragma unroll
    for (int t = 0; t < 27; ++t) s += ov[t] * wc[t];
    out[base + c] = s + x[base + c];
  }
}

// ---------------- launcher ----------------------------------------------
extern "C" void kernel_launch(void* const* d_in, const int* in_sizes, int n_in,
                              void* d_out, int out_size, void* d_ws, size_t ws_size,
                              hipStream_t stream) {
  const float* x    = (const float*)d_in[0];
  const float* cent = (const float*)d_in[1];
  const float* dwcw = (const float*)d_in[2];
  const float* dwcb = (const float*)d_in[3];
  const float* upcw = (const float*)d_in[4];
  const float* upcb = (const float*)d_in[5];
  const float* qw   = (const float*)d_in[6];
  const float* qb   = (const float*)d_in[7];
  const float* kvw  = (const float*)d_in[8];
  const float* kvb  = (const float*)d_in[9];
  float* out = (float*)d_out;

  char* p = (char*)d_ws;
  float*    fea_raw = (float*)p;    p += (size_t)NTOK * FEAD * 4;
  _Float16* feah    = (_Float16*)p; p += (size_t)NTOK * FEAD * 2;
  float*    logits  = (float*)p;    p += (size_t)NTOK * KCL * 4;
  _Float16* Et      = (_Float16*)p; p += (size_t)KCL * NTOK * 2;
  _Float16* qh      = (_Float16*)p; p += (size_t)NTOK * FEAD * 2;
  float*    ogrid   = (float*)p;    p += (size_t)32 * 128 * 128 * 4;
  float*    ncen    = (float*)p;    p += (size_t)KCL * FEAD * 4;
  _Float16* kth     = (_Float16*)p; p += (size_t)FEAD * KCL * 2;
  _Float16* vh      = (_Float16*)p; p += (size_t)KCL * FEAD * 2;
  float*    cn      = (float*)p;    p += 256;
  float*    cmax    = (float*)p;    p += 256;
  float*    csum    = (float*)p;    p += 256;

  k_dwc<<<2048, 256, 0, stream>>>(x, dwcw, dwcb, fea_raw);
  k_colnorm<<<64, 256, 0, stream>>>(fea_raw, cn);
  k_feanorm<<<2048, 256, 0, stream>>>(fea_raw, cn, feah);
  k_logits<<<dim3(64, 4), 256, 0, stream>>>(feah, cent, logits);
  k_colmaxsum<<<64, 256, 0, stream>>>(logits, cmax, csum);
  k_expT<<<2048, 256, 0, stream>>>(logits, cmax, Et);
  k_newcent<<<1, 512, 0, stream>>>(Et, feah, csum, ncen);
  k_kv<<<1, 256, 0, stream>>>(ncen, kvw, kvb, kth, vh);
  k_q<<<dim3(64, 4), 256, 0, stream>>>(feah, qw, qb, qh);
  k_attn<<<64, 256, 0, stream>>>(qh, kth, vh, ogrid);
  k_upc<<<2048, 256, 0, stream>>>(ogrid, upcw, upcb, x, out);
}
